// ComplexityAwareAttention_26534307954895
// MI455X (gfx1250) — matrix operand order repaired and checked by reading, not compiled, not run on hardware
//
#include <hip/hip_runtime.h>
#include <hip/hip_bf16.h>

// ---------------------------------------------------------------------------
// ComplexityAwareAttention for gfx1250 (MI455X), bf16 WMMA + f32 accumulate,
// TDM async-tensor staging for the attention V tile.
// B=2, S=2048, D=1024, H=16, DK=64.
// ---------------------------------------------------------------------------

typedef __bf16 bf16_t;
typedef __attribute__((ext_vector_type(16))) __bf16 v16bf;
typedef __attribute__((ext_vector_type(8)))  float  v8f;
typedef __attribute__((ext_vector_type(4)))  unsigned int u32x4;
typedef __attribute__((ext_vector_type(8)))  int  i32x8;
typedef __attribute__((ext_vector_type(4)))  int  i32x4;

#define CAA_B  2
#define CAA_S  2048
#define CAA_D  1024
#define CAA_H  16
#define CAA_DK 64
#define CAA_M  (CAA_B * CAA_S)   // 4096 rows

#if defined(__has_builtin)
#if __has_builtin(__builtin_amdgcn_tensor_load_to_lds) && \
    __has_builtin(__builtin_amdgcn_s_wait_tensorcnt)
#define CAA_HAS_TDM 1
#endif
#endif

union frag_u  { v16bf v; uint4 q[2]; };
union pack_u  { uint4 q; unsigned short h[8]; };

static __device__ __forceinline__ v8f wmma_bf16(v16bf a, v16bf b, v8f c) {
    return __builtin_amdgcn_wmma_f32_16x16x32_bf16(
        /*neg_a=*/false, a, /*neg_b=*/false, b,
        /*c_mod=*/(short)0, c, /*reuse_a=*/false, /*reuse_b=*/false);
}

#ifdef CAA_HAS_TDM
// TDM: async-load a 2D tile [rows x cols] of bf16 (row stride = strideElems)
// from global memory into LDS at ldsByteAddr.  D# per CDNA5 ISA ch.8.
static __device__ __forceinline__ void tdm_load_tile_bf16(
    unsigned int ldsByteAddr, const bf16_t* gptr,
    int cols, int rows, int strideElems)
{
    unsigned long long ga = (unsigned long long)(const void*)gptr;
    u32x4 g0;
    g0[0] = 1u;                                   // count=1, user desc
    g0[1] = ldsByteAddr;                          // lds_addr (bytes)
    g0[2] = (unsigned int)(ga & 0xFFFFFFFFu);     // global_addr[31:0]
    g0[3] = (unsigned int)((ga >> 32) & 0x01FFFFFFu) | (2u << 30); // [56:32]|type=2
    i32x8 g1;
    g1[0] = 1 << 16;                              // data_size=1 (2 bytes)
    g1[1] = (cols & 0xFFFF) << 16;                // tensor_dim0[15:0] @bits63:48
    g1[2] = (rows & 0xFFFF) << 16;                // tensor_dim1[15:0] @bits95:80
    g1[3] = (cols & 0xFFFF) << 16;                // tile_dim0 @bits127:112
    g1[4] = rows & 0xFFFF;                        // tile_dim1 @bits143:128
    g1[5] = strideElems;                          // tensor_dim0_stride[31:0]
    g1[6] = 0;
    g1[7] = 0;
    i32x4 gz = (i32x4){0, 0, 0, 0};
#if defined(__clang_major__) && __clang_major__ >= 23
    i32x8 gz8 = (i32x8){0, 0, 0, 0, 0, 0, 0, 0};
    __builtin_amdgcn_tensor_load_to_lds(g0, g1, gz, gz, gz8, 0);
#else
    __builtin_amdgcn_tensor_load_to_lds(g0, g1, gz, gz, 0);
#endif
}
#endif

// ---------------------------------------------------------------------------
// GEMM: C[M x N] = A[M x K] * W[N x K]^T (+ bias) (+ residual)
// A is f32 (Af32) or bf16 (Abf). Output bf16 (Cb) or f32 (Cf).
// Block: 128 threads = 4 waves. Block tile 128(M) x 64(N), K chunk 32.
// Each wave: two 16-row strips x four 16-col tiles -> 8 WMMA per K chunk.
// ---------------------------------------------------------------------------
__global__ __launch_bounds__(128) void caa_gemm_bf16(
    const float* __restrict__ Af32, const bf16_t* __restrict__ Abf,
    const float* __restrict__ Wf,   const float* __restrict__ bias,
    const float* __restrict__ residual,
    bf16_t* __restrict__ Cb, float* __restrict__ Cf,
    int M, int N, int K)
{
    __shared__ __align__(64) bf16_t As[128][32];   // [m][k]  8KB
    __shared__ __align__(64) bf16_t Bs[32][64];    // [k][n]  4KB (transposed)

    const int tid  = threadIdx.x;
    const int w    = tid >> 5;
    const int lane = tid & 31;
    const int hh   = lane >> 4;
    const int ml   = lane & 15;

    const int blockN = blockIdx.x * 64;
    const int blockM = blockIdx.y * 128;

    // B staging: thread owns an adjacent N-row *pair* and an 8-wide K segment,
    // so transposed stores are packed 32-bit (2 bf16 of adjacent n per store).
    const int bn0  = (tid >> 2) * 2;   // 0,2,..,62
    const int bks  = (tid & 3) * 8;    // 0,8,16,24

    v8f acc[2][4];
    #pragma unroll
    for (int s = 0; s < 2; ++s)
        #pragma unroll
        for (int t = 0; t < 4; ++t) acc[s][t] = (v8f){0,0,0,0,0,0,0,0};

    for (int kc = 0; kc < K; kc += 32) {
        // ---- stage A tile [128 x 32]: thread handles one full row ----
        {
            size_t off = (size_t)(blockM + tid) * K + kc;
            if (Af32) {
                const float* src = Af32 + off;
                __builtin_prefetch(src + 32, 0, 1);
                #pragma unroll
                for (int j = 0; j < 32; ++j) As[tid][j] = (bf16_t)src[j];
            } else {
                const bf16_t* src = Abf + off;
                __builtin_prefetch(src + 32, 0, 1);
                #pragma unroll
                for (int j = 0; j < 4; ++j)
                    *(uint4*)&As[tid][j * 8] = *(const uint4*)(src + j * 8);
            }
        }
        // ---- stage W tile transposed, packed pairs: Bs[k][n0..n0+1] ----
        {
            size_t off = (size_t)(blockN + bn0) * K + kc + bks;
            const float* s0 = Wf + off;
            const float* s1 = s0 + K;
            __builtin_prefetch(s0 + 32, 0, 1);
            #pragma unroll
            for (int j = 0; j < 8; ++j) {
                union { bf16_t h[2]; unsigned int u; } pk;
                pk.h[0] = (bf16_t)s0[j];
                pk.h[1] = (bf16_t)s1[j];
                *(unsigned int*)&Bs[bks + j][bn0] = pk.u;
            }
        }
        __syncthreads();

        // ---- A fragments: two 16-row strips, 2x uint4 per lane ----
        frag_u af[2];
        #pragma unroll
        for (int s = 0; s < 2; ++s) {
            const int mr = w * 32 + s * 16 + ml;
            af[s].q[0] = *(const uint4*)&As[mr][8 * hh];
            af[s].q[1] = *(const uint4*)&As[mr][16 + 8 * hh];
        }
        // ---- B fragments: lane = N column, 16 K elements in table order ----
        #pragma unroll
        for (int t = 0; t < 4; ++t) {
            v16bf bfrag;
            #pragma unroll
            for (int e = 0; e < 16; ++e)
                bfrag[e] = Bs[8 * hh + (e & 7) + 16 * (e >> 3)][t * 16 + ml];
            acc[0][t] = wmma_bf16(af[0].v, bfrag, acc[0][t]);
            acc[1][t] = wmma_bf16(af[1].v, bfrag, acc[1][t]);
        }
        __syncthreads();
    }

    // ---- epilogue: C layout row m = r + 8*hh, col n = ml ----
    #pragma unroll
    for (int s = 0; s < 2; ++s) {
        #pragma unroll
        for (int t = 0; t < 4; ++t) {
            #pragma unroll
            for (int r = 0; r < 8; ++r) {
                const int row = blockM + w * 32 + s * 16 + r + 8 * hh;
                const int col = blockN + t * 16 + ml;
                float v = acc[s][t][r] + bias[col];
                size_t idx = (size_t)row * N + col;
                if (Cf) {
                    if (residual) v += residual[idx];
                    Cf[idx] = v;
                } else {
                    Cb[idx] = (bf16_t)v;
                }
            }
        }
    }
}

// ---------------------------------------------------------------------------
// Flash attention: one block per (b, h, 64 query rows). 4 waves x 16 q-rows.
// V tile: TDM async load to LDS (natural [key][dk] layout).
// K tile: manual transpose into Kt[dk][key] with packed 32-bit stores.
// ---------------------------------------------------------------------------
__global__ __launch_bounds__(128) void caa_attention(
    const bf16_t* __restrict__ Qp, const bf16_t* __restrict__ Kp,
    const bf16_t* __restrict__ Vp, const float* __restrict__ comp,
    const float* __restrict__ cpen_p, bf16_t* __restrict__ ctxb)
{
    __shared__ __align__(64) bf16_t Kt[CAA_DK][32];   // [dk][key] 4KB
    __shared__ __align__(64) bf16_t Vs[32][CAA_DK];   // [key][dk] 4KB
    __shared__ __align__(64) bf16_t Ps[4][16][32];    // per-wave P 4KB

    const int tid  = threadIdx.x;
    const int w    = tid >> 5;
    const int lane = tid & 31;
    const int hh   = lane >> 4;
    const int ml   = lane & 15;

    const int idx  = blockIdx.x;
    const int qblk = idx & 31;               // S/64 = 32
    const int h    = (idx >> 5) & 15;
    const int b    = idx >> 9;

    const int q0b = qblk * 64;               // block query base
    const int q0  = q0b + w * 16;            // wave query base

    const float cpen = cpen_p[0];

    // ---- Q fragments (DK=64 -> 2 chunks of K32), loaded once ----
    frag_u qa[2];
    {
        const bf16_t* qrow = Qp + ((size_t)(b * CAA_S + q0 + ml)) * CAA_D + h * CAA_DK;
        #pragma unroll
        for (int c = 0; c < 2; ++c) {
            qa[c].q[0] = *(const uint4*)(qrow + c * 32 + 8 * hh);
            qa[c].q[1] = *(const uint4*)(qrow + c * 32 + 16 + 8 * hh);
        }
    }

    v8f o[4];
    #pragma unroll
    for (int t = 0; t < 4; ++t) o[t] = (v8f){0,0,0,0,0,0,0,0};
    float Mrow[8], Lrow[8];
    #pragma unroll
    for (int r = 0; r < 8; ++r) { Mrow[r] = -1e30f; Lrow[r] = 0.0f; }

    // K staging: thread owns a key *pair* and an 8-wide dk segment.
    const int key0 = (tid >> 3) * 2;     // 0,2,..,30
    const int dseg = (tid & 7) * 8;      // 0,8,..,56

    const int tend = q0b + 64;           // causal bound for the block
    for (int t0 = 0; t0 < tend; t0 += 32) {
        const size_t tileOff = ((size_t)(b * CAA_S + t0)) * CAA_D + h * CAA_DK;

#ifdef CAA_HAS_TDM
        // ---- V tile via Tensor Data Mover (issued once, overlaps with K) ----
        if (w == 0) {
            tdm_load_tile_bf16((unsigned int)(size_t)&Vs[0][0], Vp + tileOff,
                               CAA_DK, 32, CAA_D);
        }
#endif
        // ---- K tile transposed: packed 32-bit stores (2 adjacent keys) ----
        {
            const size_t offA = tileOff + (size_t)key0 * CAA_D + dseg;
            pack_u r0, r1;
            r0.q = *(const uint4*)(Kp + offA);
            r1.q = *(const uint4*)(Kp + offA + CAA_D);
            #pragma unroll
            for (int j = 0; j < 8; ++j) {
                *(unsigned int*)&Kt[dseg + j][key0] =
                    (unsigned int)r0.h[j] | ((unsigned int)r1.h[j] << 16);
            }
#ifndef CAA_HAS_TDM
            // fallback: V tile via vector copies (natural layout)
            *(uint4*)&Vs[key0][dseg]     = *(const uint4*)(Vp + offA);
            *(uint4*)&Vs[key0 + 1][dseg] = *(const uint4*)(Vp + offA + CAA_D);
#endif
        }
#ifdef CAA_HAS_TDM
        if (w == 0) __builtin_amdgcn_s_wait_tensorcnt(0);
#endif
        __syncthreads();

        if (t0 <= q0 + 15) {   // tile intersects this wave's causal window
            // ---- scores: Q(16x64) x K^T(64x32) -> two 16x16 tiles ----
            v8f s0 = (v8f){0,0,0,0,0,0,0,0};
            v8f s1 = (v8f){0,0,0,0,0,0,0,0};
            #pragma unroll
            for (int c = 0; c < 2; ++c) {
                v16bf b0, b1;   // lane = key column, elements = dk (table order)
                #pragma unroll
                for (int e = 0; e < 16; ++e) {
                    const int dk = c * 32 + 8 * hh + (e & 7) + 16 * (e >> 3);
                    b0[e] = Kt[dk][ml];
                    b1[e] = Kt[dk][16 + ml];
                }
                s0 = wmma_bf16(qa[c].v, b0, s0);
                s1 = wmma_bf16(qa[c].v, b1, s1);
            }

            // ---- online softmax (rows live in 16-lane halves) ----
            const float pen0 = cpen * comp[b * CAA_S + t0 + ml];
            const float pen1 = cpen * comp[b * CAA_S + t0 + 16 + ml];
            const int tn0 = t0 + ml, tn1 = t0 + 16 + ml;
            #pragma unroll
            for (int r = 0; r < 8; ++r) {
                const int q = q0 + r + 8 * hh;
                float sc0 = s0[r] * 0.125f - pen0;
                float sc1 = s1[r] * 0.125f - pen1;
                if (tn0 > q) sc0 = -1e9f;
                if (tn1 > q) sc1 = -1e9f;
                float tm = fmaxf(sc0, sc1);
                #pragma unroll
                for (int off = 1; off < 16; off <<= 1)
                    tm = fmaxf(tm, __shfl_xor(tm, off, 16));
                const float Mn = fmaxf(Mrow[r], tm);
                const float p0 = __expf(sc0 - Mn);
                const float p1 = __expf(sc1 - Mn);
                float rs = p0 + p1;
                #pragma unroll
                for (int off = 1; off < 16; off <<= 1)
                    rs += __shfl_xor(rs, off, 16);
                const float scale = __expf(Mrow[r] - Mn);
                Lrow[r] = Lrow[r] * scale + rs;
                Mrow[r] = Mn;
                #pragma unroll
                for (int t = 0; t < 4; ++t) o[t][r] *= scale;
                Ps[w][r + 8 * hh][ml]      = (bf16_t)p0;
                Ps[w][r + 8 * hh][16 + ml] = (bf16_t)p1;
            }

            // ---- P(16x32) x V(32x64) -> 4 ctx tiles ----
            frag_u pa;
            pa.q[0] = *(const uint4*)&Ps[w][ml][8 * hh];
            pa.q[1] = *(const uint4*)&Ps[w][ml][16 + 8 * hh];
            #pragma unroll
            for (int t = 0; t < 4; ++t) {
                v16bf vb;   // lane = dk column, elements = key (table order)
                #pragma unroll
                for (int e = 0; e < 16; ++e)
                    vb[e] = Vs[8 * hh + (e & 7) + 16 * (e >> 3)][t * 16 + ml];
                o[t] = wmma_bf16(pa.v, vb, o[t]);
            }
        }
        __syncthreads();
    }

    // ---- normalize and write ctx (bf16, [B*S, D] with head offset) ----
    #pragma unroll
    for (int r = 0; r < 8; ++r) {
        const float inv = 1.0f / Lrow[r];
        const int row = b * CAA_S + q0 + r + 8 * hh;
        #pragma unroll
        for (int t = 0; t < 4; ++t) {
            size_t gi = (size_t)row * CAA_D + h * CAA_DK + t * 16 + ml;
            ctxb[gi] = (bf16_t)(o[t][r] * inv);
        }
    }
}

// ---------------------------------------------------------------------------
// LayerNorm over rows of 1024, one block (256 threads) per row.
// ---------------------------------------------------------------------------
__global__ __launch_bounds__(256) void caa_layernorm(
    const float* __restrict__ x, const float* __restrict__ g,
    const float* __restrict__ bt, float* __restrict__ out)
{
    __shared__ float rs[256];
    __shared__ float rs2[256];
    const int row = blockIdx.x;
    const int tid = threadIdx.x;
    const float* xr = x + (size_t)row * CAA_D;

    float s = 0.0f, s2 = 0.0f;
    #pragma unroll
    for (int i = tid; i < CAA_D; i += 256) {
        float v = xr[i];
        s += v; s2 += v * v;
    }
    rs[tid] = s; rs2[tid] = s2;
    __syncthreads();
    for (int off = 128; off > 0; off >>= 1) {
        if (tid < off) { rs[tid] += rs[tid + off]; rs2[tid] += rs2[tid + off]; }
        __syncthreads();
    }
    const float mean = rs[0] * (1.0f / CAA_D);
    const float var  = rs2[0] * (1.0f / CAA_D) - mean * mean;
    const float inv  = rsqrtf(var + 1e-5f);
    float* orow = out + (size_t)row * CAA_D;
    #pragma unroll
    for (int i = tid; i < CAA_D; i += 256)
        orow[i] = (xr[i] - mean) * inv * g[i] + bt[i];
}

// ---------------------------------------------------------------------------
// Launcher
// ---------------------------------------------------------------------------
extern "C" void kernel_launch(void* const* d_in, const int* in_sizes, int n_in,
                              void* d_out, int out_size, void* d_ws, size_t ws_size,
                              hipStream_t stream) {
    const float* query = (const float*)d_in[0];
    const float* keyi  = (const float*)d_in[1];
    const float* value = (const float*)d_in[2];
    const float* comp  = (const float*)d_in[3];
    // d_in[4] = dense causal mask (unused; causality computed analytically)
    const float* wq = (const float*)d_in[5];
    const float* bq = (const float*)d_in[6];
    const float* wk = (const float*)d_in[7];
    const float* bk = (const float*)d_in[8];
    const float* wv = (const float*)d_in[9];
    const float* bv = (const float*)d_in[10];
    const float* wo = (const float*)d_in[11];
    const float* bo = (const float*)d_in[12];
    const float* ln_g = (const float*)d_in[13];
    const float* ln_b = (const float*)d_in[14];
    const float* cpen = (const float*)d_in[15];
    float* out = (float*)d_out;

    // workspace carve-up (~48 MB)
    char* p = (char*)d_ws;
    const size_t MD = (size_t)CAA_M * CAA_D;
    bf16_t* Qp   = (bf16_t*)p; p += MD * sizeof(bf16_t);
    bf16_t* Kp   = (bf16_t*)p; p += MD * sizeof(bf16_t);
    bf16_t* Vp   = (bf16_t*)p; p += MD * sizeof(bf16_t);
    bf16_t* ctxb = (bf16_t*)p; p += MD * sizeof(bf16_t);
    float*  xbuf = (float*)p;  p += MD * sizeof(float);

    dim3 gblk(128);
    dim3 ggrid(CAA_D / 64, CAA_M / 128);   // (16, 32)

    // Q/K/V projections (f32 A -> bf16 out)
    caa_gemm_bf16<<<ggrid, gblk, 0, stream>>>(query, nullptr, wq, bq, nullptr,
                                              Qp, nullptr, CAA_M, CAA_D, CAA_D);
    caa_gemm_bf16<<<ggrid, gblk, 0, stream>>>(keyi, nullptr, wk, bk, nullptr,
                                              Kp, nullptr, CAA_M, CAA_D, CAA_D);
    caa_gemm_bf16<<<ggrid, gblk, 0, stream>>>(value, nullptr, wv, bv, nullptr,
                                              Vp, nullptr, CAA_M, CAA_D, CAA_D);

    // flash attention: B*H*(S/64) = 1024 blocks
    caa_attention<<<dim3(CAA_B * CAA_H * (CAA_S / 64)), gblk, 0, stream>>>(
        Qp, Kp, Vp, comp, cpen, ctxb);

    // output projection + bias + residual (bf16 A -> f32 out)
    caa_gemm_bf16<<<ggrid, gblk, 0, stream>>>(nullptr, ctxb, wo, bo, query,
                                              nullptr, xbuf, CAA_M, CAA_D, CAA_D);

    // layernorm
    caa_layernorm<<<dim3(CAA_M), dim3(256), 0, stream>>>(xbuf, ln_g, ln_b, out);
}
